// KVCacheMultiHeadAttention_48576080118369
// MI455X (gfx1250) — matrix operand order repaired and checked by reading, not compiled, not run on hardware
//
#include <hip/hip_runtime.h>

// Problem constants (match reference)
#define DMODEL 2048
#define NHEADS 16
#define HDIM   128
#define BATCH  4
#define SEQ    2048
#define NTOK   (BATCH * SEQ)   // 8192 tokens

typedef __bf16 bf16;
typedef __attribute__((ext_vector_type(16))) __bf16 v16bf;
typedef __attribute__((ext_vector_type(8)))  __bf16 v8bf;
typedef __attribute__((ext_vector_type(8)))  float  v8f;

union FragAB { v16bf v; v8bf h[2]; };

#if __has_builtin(__builtin_amdgcn_tensor_load_to_lds) && \
    __has_builtin(__builtin_amdgcn_s_wait_tensorcnt)
#define HAVE_TDM 1
#else
#define HAVE_TDM 0
#endif

static __device__ __forceinline__ bf16 f2bf(float f) {
  union { float f; unsigned u; } in; in.f = f;
  unsigned r = in.u + 0x7FFFu + ((in.u >> 16) & 1u);   // round-nearest-even
  union { unsigned short s; bf16 b; } out; out.s = (unsigned short)(r >> 16);
  return out.b;
}

static __device__ __forceinline__ v8f wmma_bf16(v16bf a, v16bf b, v8f c) {
  // D = A(16x32) * B(32x16) + C, fp32 accumulate
  return __builtin_amdgcn_wmma_f32_16x16x32_bf16(false, a, false, b, (short)0, c,
                                                 false, false);
}

#if HAVE_TDM
typedef __attribute__((ext_vector_type(4))) unsigned int tdm_u32x4;
typedef __attribute__((ext_vector_type(8))) int          tdm_i32x8;
typedef __attribute__((ext_vector_type(4))) int          tdm_i32x4;

// Issue one TDM 2D tile load: global (row-major, bf16) -> LDS, with TDM-side
// row padding. Descriptor layout per CDNA5 ISA ch.8 (D# groups 0/1).
// Toolchain exposes the 6-arg builtin: (g0, g1, g2, g3, g4, cpol).
static __device__ __forceinline__ void tdm_load_tile_2d(
    unsigned lds_byte_addr, const void* global_tile,
    unsigned tile_elems_d0,               // elements per row (bf16)
    unsigned tile_rows,                   // rows in tile
    unsigned long long row_stride_elems,  // tensor_dim0_stride (elements)
    unsigned pad_interval_code,           // 3 => every 16 DWORDs, 5 => every 64
    unsigned pad_amount_code) {           // 3 => pad 4 DWORDs (16B)
  unsigned long long ga = (unsigned long long)global_tile;
  unsigned td0 = (unsigned)row_stride_elems;   // tensor_dim0 (OOB never hit)
  unsigned td1 = 1u << 20;                     // tensor_dim1 (large)

  tdm_u32x4 g0;
  g0[0] = 1u;                                       // count=1, user descriptor
  g0[1] = lds_byte_addr;                            // lds_addr
  g0[2] = (unsigned)ga;                             // global_addr[31:0]
  g0[3] = (unsigned)((ga >> 32) & 0x01FFFFFFu) | (2u << 30);  // addr[56:32]|type=2

  tdm_i32x8 g1;
  g1[0] = (int)((1u << 16) |                        // data_size = 2 bytes
                (1u << 20) |                        // pad_enable
                (pad_interval_code << 22) | (pad_amount_code << 25));
  g1[1] = (int)((td0 & 0xFFFFu) << 16);             // tensor_dim0[15:0]
  g1[2] = (int)(((td0 >> 16) & 0xFFFFu) | ((td1 & 0xFFFFu) << 16));
  g1[3] = (int)(((td1 >> 16) & 0xFFFFu) | (tile_elems_d0 << 16));  // tile_dim0
  g1[4] = (int)(tile_rows & 0xFFFFu);               // tile_dim1 (tile_dim2=0)
  g1[5] = (int)(unsigned)(row_stride_elems & 0xFFFFFFFFull);
  g1[6] = (int)(unsigned)((row_stride_elems >> 32) & 0xFFFFull);
  g1[7] = 0;

  tdm_i32x4 z4 = {0, 0, 0, 0};                      // groups 2/3 unused (2D)
  tdm_i32x8 z8 = {0, 0, 0, 0, 0, 0, 0, 0};
  __builtin_amdgcn_tensor_load_to_lds(g0, g1, z4, z4, z8, 0);
}
#endif

// ---------------------------------------------------------------------------
// fp32 -> bf16 bulk convert (n multiple of 8)
// ---------------------------------------------------------------------------
__global__ __launch_bounds__(256)
void cvt_f32_bf16(const float* __restrict__ s, bf16* __restrict__ d, int n) {
  int i = (blockIdx.x * 256 + threadIdx.x) * 8;
  if (i + 8 <= n) {
    float4 f0 = *(const float4*)(s + i);
    float4 f1 = *(const float4*)(s + i + 4);
    v8bf p;
    p[0] = f2bf(f0.x); p[1] = f2bf(f0.y); p[2] = f2bf(f0.z); p[3] = f2bf(f0.w);
    p[4] = f2bf(f1.x); p[5] = f2bf(f1.y); p[6] = f2bf(f1.z); p[7] = f2bf(f1.w);
    *(v8bf*)(d + i) = p;
  }
}

// ---------------------------------------------------------------------------
// GEMM:  C[M,N] = A[M,K](bf16) * W[N,K]^T(bf16) + bias[N](f32)
// Tile 128x128, BK=32 double-buffered via TDM; 256 threads = 8 waves
// (2 M-strips x 4 N-strips), each wave owns 64x32 = 4x2 WMMA tiles.
// ---------------------------------------------------------------------------
template <bool OUT_F32>
__global__ __launch_bounds__(256)
void gemm_bias_kernel(const bf16* __restrict__ A, const bf16* __restrict__ W,
                      const float* __restrict__ bias, void* __restrict__ Cout,
                      int M, int N, int K) {
  __shared__ bf16 As[2][128][40];   // 32 K + 8 pad -> 80B row stride (16B mult)
  __shared__ bf16 Bs[2][128][40];

  const int n0   = blockIdx.x * 128;
  const int m0   = blockIdx.y * 128;
  const int lane = threadIdx.x & 31;
  const int wid  = threadIdx.x >> 5;
  const int waveM = wid & 1;     // 0..1  (64-row strip)
  const int waveN = wid >> 1;    // 0..3  (32-col strip)
  const int half  = lane >> 4;   // which 16-lane half
  const int l16   = lane & 15;

  v8f acc[4][2];
  v8f zero = {0.f, 0.f, 0.f, 0.f, 0.f, 0.f, 0.f, 0.f};
#pragma unroll
  for (int mi = 0; mi < 4; ++mi)
#pragma unroll
    for (int ni = 0; ni < 2; ++ni) acc[mi][ni] = zero;

  const int nk = K / 32;
#if HAVE_TDM
  if (wid == 0) {   // wave 0 is the DMA driver
    tdm_load_tile_2d((unsigned)(size_t)&As[0][0][0], A + (size_t)m0 * K,
                     32u, 128u, (unsigned long long)K, 3u, 3u);
    tdm_load_tile_2d((unsigned)(size_t)&Bs[0][0][0], W + (size_t)n0 * K,
                     32u, 128u, (unsigned long long)K, 3u, 3u);
  }
#endif

  for (int t = 0; t < nk; ++t) {
    const int buf = t & 1;
#if HAVE_TDM
    if (wid == 0) {
      if (t + 1 < nk) {   // prefetch next tile into other buffer via TDM
        tdm_load_tile_2d((unsigned)(size_t)&As[buf ^ 1][0][0],
                         A + (size_t)m0 * K + (size_t)(t + 1) * 32,
                         32u, 128u, (unsigned long long)K, 3u, 3u);
        tdm_load_tile_2d((unsigned)(size_t)&Bs[buf ^ 1][0][0],
                         W + (size_t)n0 * K + (size_t)(t + 1) * 32,
                         32u, 128u, (unsigned long long)K, 3u, 3u);
        __builtin_amdgcn_s_wait_tensorcnt(2);   // tile t resident
      } else {
        __builtin_amdgcn_s_wait_tensorcnt(0);
      }
    }
#else
    // synchronous fallback staging of tile t
#pragma unroll
    for (int i = 0; i < 2; ++i) {
      int lid = threadIdx.x + i * 256;    // 512 chunks of 8 bf16
      int row = lid >> 2;
      int cc  = (lid & 3) * 8;
      *(v8bf*)&As[buf][row][cc] =
          *(const v8bf*)(A + (size_t)(m0 + row) * K + t * 32 + cc);
      *(v8bf*)&Bs[buf][row][cc] =
          *(const v8bf*)(W + (size_t)(n0 + row) * K + t * 32 + cc);
    }
#endif
    __syncthreads();

    FragAB a[4], b[2];
#pragma unroll
    for (int mi = 0; mi < 4; ++mi) {
      int row = waveM * 64 + mi * 16 + l16;
      int k0  = half * 8;                  // A layout: {k0..k0+7, k0+16..k0+23}
      a[mi].h[0] = *(const v8bf*)&As[buf][row][k0];
      a[mi].h[1] = *(const v8bf*)&As[buf][row][k0 + 16];
    }
#pragma unroll
    for (int ni = 0; ni < 2; ++ni) {
      int row = waveN * 32 + ni * 16 + l16;
      int k0  = half * 8;                  // B layout: {k0..k0+7, k0+16..k0+23}
      b[ni].h[0] = *(const v8bf*)&Bs[buf][row][k0];
      b[ni].h[1] = *(const v8bf*)&Bs[buf][row][k0 + 16];
    }
#pragma unroll
    for (int mi = 0; mi < 4; ++mi)
#pragma unroll
      for (int ni = 0; ni < 2; ++ni)
        acc[mi][ni] = wmma_bf16(a[mi].v, b[ni].v, acc[mi][ni]);
    __syncthreads();
  }

  // epilogue: bias add, store (C layout: VGPR r -> row r (lanes<16) / r+8)
#pragma unroll
  for (int ni = 0; ni < 2; ++ni) {
    int gn = n0 + waveN * 32 + ni * 16 + l16;
    float bv = bias[gn];
#pragma unroll
    for (int mi = 0; mi < 4; ++mi) {
#pragma unroll
      for (int r = 0; r < 8; ++r) {
        int gm = m0 + waveM * 64 + mi * 16 + r + 8 * half;
        float val = acc[mi][ni][r] + bv;
        if constexpr (OUT_F32)
          ((float*)Cout)[(size_t)gm * N + gn] = val;
        else
          ((bf16*)Cout)[(size_t)gm * N + gn] = f2bf(val);
      }
    }
  }
}

// ---------------------------------------------------------------------------
// Flash attention (no mask). Grid: (T/64, H, B). Block = 128 threads = 4 waves.
// Each wave owns a 16-query strip; K-tile = 64 keys per iteration.
// K tile staged by TDM (with TDM-side padding), V staged transposed manually.
// ---------------------------------------------------------------------------
__global__ __launch_bounds__(128)
void attn_kernel(const bf16* __restrict__ Q, const bf16* __restrict__ Kmat,
                 const bf16* __restrict__ V, bf16* __restrict__ O) {
  __shared__ bf16 Ks[64][136];    // 64 keys x 128 d (pad 8) -> 272B stride
  __shared__ bf16 Vt[128][72];    // transposed: d x 64 keys (pad 8)
  __shared__ bf16 Ps[4][16][72];  // per-wave P strip 16x64 (pad 8)

  const int q0    = blockIdx.x * 64;
  const int h     = blockIdx.y;
  const int bbase = blockIdx.z * SEQ;
  const int lane  = threadIdx.x & 31;
  const int w     = threadIdx.x >> 5;
  const int half  = lane >> 4;
  const int l16   = lane & 15;
  const float scale = 0.088388347648318447f;  // 1/sqrt(128)

  // Q fragments: persistent in registers (4 x 16x32 along Dh=128)
  FragAB qa[4];
  {
    const bf16* qptr = Q + (size_t)(bbase + q0 + w * 16 + l16) * DMODEL + h * HDIM;
#pragma unroll
    for (int ks = 0; ks < 4; ++ks) {
      int k0 = ks * 32 + half * 8;
      qa[ks].h[0] = *(const v8bf*)(qptr + k0);
      qa[ks].h[1] = *(const v8bf*)(qptr + k0 + 16);
    }
  }

  v8f zero = {0.f, 0.f, 0.f, 0.f, 0.f, 0.f, 0.f, 0.f};
  v8f o[8];
#pragma unroll
  for (int n = 0; n < 8; ++n) o[n] = zero;
  float m[8], l[8];
#pragma unroll
  for (int r = 0; r < 8; ++r) { m[r] = -1e30f; l[r] = 0.f; }

  for (int kt = 0; kt < SEQ / 64; ++kt) {
    const int kk0 = kt * 64;
#if HAVE_TDM
    if (w == 0)   // DMA the K tile (TDM pads rows 256B -> 272B)
      tdm_load_tile_2d((unsigned)(size_t)&Ks[0][0],
                       Kmat + (size_t)(bbase + kk0) * DMODEL + h * HDIM,
                       (unsigned)HDIM, 64u, (unsigned long long)DMODEL, 5u, 3u);
    // all waves stage V transposed meanwhile
#pragma unroll
    for (int i = 0; i < 8; ++i) {
      int lid = threadIdx.x + i * 128;       // 1024 chunks of 8 bf16
      int row = lid >> 4;                    // key row 0..63
      int cc  = (lid & 15) * 8;              // d offset
      v8bf vv =
          *(const v8bf*)(V + (size_t)(bbase + kk0 + row) * DMODEL + h * HDIM + cc);
#pragma unroll
      for (int j = 0; j < 8; ++j) Vt[cc + j][row] = vv[j];
    }
    if (w == 0) __builtin_amdgcn_s_wait_tensorcnt(0);
#else
#pragma unroll
    for (int i = 0; i < 8; ++i) {
      int lid = threadIdx.x + i * 128;
      int row = lid >> 4;
      int cc  = (lid & 15) * 8;
      *(v8bf*)&Ks[row][cc] =
          *(const v8bf*)(Kmat + (size_t)(bbase + kk0 + row) * DMODEL + h * HDIM + cc);
      v8bf vv =
          *(const v8bf*)(V + (size_t)(bbase + kk0 + row) * DMODEL + h * HDIM + cc);
#pragma unroll
      for (int j = 0; j < 8; ++j) Vt[cc + j][row] = vv[j];
    }
#endif
    __syncthreads();

    // S = Q * K^T  (4 n-tiles x 4 k-steps)
    v8f s[4];
#pragma unroll
    for (int n = 0; n < 4; ++n) s[n] = zero;
#pragma unroll
    for (int ks = 0; ks < 4; ++ks) {
#pragma unroll
      for (int n = 0; n < 4; ++n) {
        FragAB b;
        int key = n * 16 + l16;
        int k0  = ks * 32 + half * 8;
        b.h[0] = *(const v8bf*)&Ks[key][k0];
        b.h[1] = *(const v8bf*)&Ks[key][k0 + 16];
        s[n] = wmma_bf16(qa[ks].v, b.v, s[n]);
      }
    }

    // streaming softmax
    float rm[8];
#pragma unroll
    for (int r = 0; r < 8; ++r) {
#pragma unroll
      for (int n = 0; n < 4; ++n) s[n][r] *= scale;
      rm[r] = fmaxf(fmaxf(s[0][r], s[1][r]), fmaxf(s[2][r], s[3][r]));
    }
#pragma unroll
    for (int off = 1; off < 16; off <<= 1)
#pragma unroll
      for (int r = 0; r < 8; ++r) rm[r] = fmaxf(rm[r], __shfl_xor(rm[r], off, 32));

    float alpha[8], rs[8];
#pragma unroll
    for (int r = 0; r < 8; ++r) {
      float mn = fmaxf(m[r], rm[r]);
      alpha[r] = __expf(m[r] - mn);
      m[r] = mn;
      rs[r] = 0.f;
    }
#pragma unroll
    for (int n = 0; n < 4; ++n)
#pragma unroll
      for (int r = 0; r < 8; ++r) {
        float p = __expf(s[n][r] - m[r]);
        s[n][r] = p;
        rs[r] += p;
      }
#pragma unroll
    for (int off = 1; off < 16; off <<= 1)
#pragma unroll
      for (int r = 0; r < 8; ++r) rs[r] += __shfl_xor(rs[r], off, 32);
#pragma unroll
    for (int r = 0; r < 8; ++r) l[r] = l[r] * alpha[r] + rs[r];
#pragma unroll
    for (int n = 0; n < 8; ++n)
#pragma unroll
      for (int r = 0; r < 8; ++r) o[n][r] *= alpha[r];

    // P: C-layout -> A-layout via per-wave LDS strip (same-wave DS is ordered)
#pragma unroll
    for (int n = 0; n < 4; ++n)
#pragma unroll
      for (int r = 0; r < 8; ++r)
        Ps[w][r + 8 * half][n * 16 + l16] = f2bf(s[n][r]);

    // O += P * V   (8 d-tiles x 2 k-steps over 64 keys)
#pragma unroll
    for (int ks2 = 0; ks2 < 2; ++ks2) {
      FragAB pa;
      int pk = ks2 * 32 + half * 8;
      pa.h[0] = *(const v8bf*)&Ps[w][l16][pk];
      pa.h[1] = *(const v8bf*)&Ps[w][l16][pk + 16];
#pragma unroll
      for (int n = 0; n < 8; ++n) {
        FragAB vb;
        int drow = n * 16 + l16;
        int vk   = ks2 * 32 + half * 8;
        vb.h[0] = *(const v8bf*)&Vt[drow][vk];
        vb.h[1] = *(const v8bf*)&Vt[drow][vk + 16];
        o[n] = wmma_bf16(pa.v, vb.v, o[n]);
      }
    }
    __syncthreads();
  }

  // normalize + store attention output (bf16, [token][h*128+d])
#pragma unroll
  for (int n = 0; n < 8; ++n)
#pragma unroll
    for (int r = 0; r < 8; ++r) {
      int qr = q0 + w * 16 + r + 8 * half;
      int dc = n * 16 + l16;
      float val = o[n][r] / l[r];
      O[(size_t)(bbase + qr) * DMODEL + h * HDIM + dc] = f2bf(val);
    }
}

// ---------------------------------------------------------------------------
// launch
// ---------------------------------------------------------------------------
extern "C" void kernel_launch(void* const* d_in, const int* in_sizes, int n_in,
                              void* d_out, int out_size, void* d_ws, size_t ws_size,
                              hipStream_t stream) {
  const float* x  = (const float*)d_in[0];
  const float* Wq = (const float*)d_in[1];
  const float* bq = (const float*)d_in[2];
  const float* Wk = (const float*)d_in[3];
  const float* bk = (const float*)d_in[4];
  const float* Wv = (const float*)d_in[5];
  const float* bv = (const float*)d_in[6];
  const float* Wo = (const float*)d_in[7];
  const float* bo = (const float*)d_in[8];

  char* ws = (char*)d_ws;
  const size_t XB = (size_t)NTOK * DMODEL * sizeof(bf16);     // 32 MB
  const size_t WB = (size_t)DMODEL * DMODEL * sizeof(bf16);   // 8 MB
  bf16* xb  = (bf16*)(ws);
  bf16* Wqb = (bf16*)(ws + XB);
  bf16* Wkb = (bf16*)(ws + XB + WB);
  bf16* Wvb = (bf16*)(ws + XB + 2 * WB);
  bf16* Wob = (bf16*)(ws + XB + 3 * WB);
  bf16* Qb  = (bf16*)(ws + XB + 4 * WB);
  bf16* Kb  = (bf16*)(ws + 2 * XB + 4 * WB);
  bf16* Vb  = (bf16*)(ws + 3 * XB + 4 * WB);
  bf16* Ab  = (bf16*)(ws + 4 * XB + 4 * WB);

  const int nX = NTOK * DMODEL;     // 16,777,216
  const int nW = DMODEL * DMODEL;   // 4,194,304
  cvt_f32_bf16<<<nX / 2048, 256, 0, stream>>>(x, xb, nX);
  cvt_f32_bf16<<<nW / 2048, 256, 0, stream>>>(Wq, Wqb, nW);
  cvt_f32_bf16<<<nW / 2048, 256, 0, stream>>>(Wk, Wkb, nW);
  cvt_f32_bf16<<<nW / 2048, 256, 0, stream>>>(Wv, Wvb, nW);
  cvt_f32_bf16<<<nW / 2048, 256, 0, stream>>>(Wo, Wob, nW);

  dim3 ggrid(DMODEL / 128, NTOK / 128);   // (16, 64)
  gemm_bias_kernel<false><<<ggrid, 256, 0, stream>>>(xb, Wqb, bq, Qb, NTOK, DMODEL, DMODEL);
  gemm_bias_kernel<false><<<ggrid, 256, 0, stream>>>(xb, Wkb, bk, Kb, NTOK, DMODEL, DMODEL);
  gemm_bias_kernel<false><<<ggrid, 256, 0, stream>>>(xb, Wvb, bv, Vb, NTOK, DMODEL, DMODEL);

  attn_kernel<<<dim3(SEQ / 64, NHEADS, BATCH), 128, 0, stream>>>(Qb, Kb, Vb, Ab);

  gemm_bias_kernel<true><<<ggrid, 256, 0, stream>>>(Ab, Wob, bo, (float*)d_out,
                                                    NTOK, DMODEL, DMODEL);
}
